// EnergyConstrainedPredictiveCodingModel_27513560498312
// MI455X (gfx1250) — matrix operand order repaired and checked by reading, not compiled, not run on hardware
//
#include <hip/hip_runtime.h>
#include <stdint.h>

// ---------------------------------------------------------------------------
// Types for CDNA5 WMMA
// ---------------------------------------------------------------------------
typedef __attribute__((ext_vector_type(16))) __bf16 v16bf;
typedef __attribute__((ext_vector_type(8)))  __bf16 v8bf;
typedef __attribute__((ext_vector_type(8)))  float  v8f;

#define B_ROWS 8192
#define D_DIM  1024
#define L_DIM  512
#define H_DIM  512
#define REC_W  256
#define OUT_W  6656

// output column offsets
#define O_Z     0
#define O_HN    512
#define O_H2N   1024
#define O_SP    1536
#define O_TH    2048
#define O_SST   2560
#define O_TFF   3072
#define O_ZE    3584
#define O_IHAT  4096
#define O_L1    5120
#define O_L2    6144

// ---------------------------------------------------------------------------
// CDNA5 async copy helpers (GLOBAL_LOAD_ASYNC_TO_LDS_*, ASYNCcnt tracked)
// ---------------------------------------------------------------------------
__device__ __forceinline__ void async_copy_b128(uint32_t lds_byte_addr, const void* gptr) {
  asm volatile("global_load_async_to_lds_b128 %0, %1, off"
               :: "v"(lds_byte_addr), "v"(gptr)
               : "memory");
}
__device__ __forceinline__ void wait_asynccnt0() {
  asm volatile("s_wait_asynccnt 0" ::: "memory");
}

// ---------------------------------------------------------------------------
// small prep kernels
// ---------------------------------------------------------------------------
__global__ void k_zero(float* p, int n) {
  int i = blockIdx.x * blockDim.x + threadIdx.x;
  if (i < n) p[i] = 0.f;
}

__global__ void k_sumsq(const float* __restrict__ w, int n, float* accum) {
  __shared__ float red[256];
  float s = 0.f;
  for (int i = blockIdx.x * blockDim.x + threadIdx.x; i < n; i += gridDim.x * blockDim.x) {
    float v = w[i]; s += v * v;
  }
  red[threadIdx.x] = s; __syncthreads();
  for (int off = 128; off > 0; off >>= 1) {
    if ((int)threadIdx.x < off) red[threadIdx.x] += red[threadIdx.x + off];
    __syncthreads();
  }
  if (threadIdx.x == 0) atomicAdd(accum, red[0]);
}

__global__ void k_cvt(__bf16* __restrict__ dst, const float* __restrict__ src, long n) {
  long i = (long)blockIdx.x * blockDim.x + threadIdx.x;
  if (i < n) dst[i] = (__bf16)src[i];
}

__global__ void k_cvt_relu(__bf16* __restrict__ dst, const float* __restrict__ src, long n) {
  long i = (long)blockIdx.x * blockDim.x + threadIdx.x;
  if (i < n) dst[i] = (__bf16)fmaxf(src[i], 0.f);
}

// dst[N x (K1+K2)] row n = [ a[n,:] | s*b[n,:] ];  s = min(1, 0.5*rsqrt(hdr[0])) if hdr
__global__ void k_wcat(__bf16* __restrict__ dst, const float* __restrict__ a,
                       const float* __restrict__ b, int N, int K1, int K2,
                       const float* hdr) {
  int K = K1 + K2;
  long total = (long)N * K;
  float s = 1.f;
  if (hdr) s = fminf(1.f, 0.5f * rsqrtf(hdr[0]));
  long i = (long)blockIdx.x * blockDim.x + threadIdx.x;
  if (i >= total) return;
  int r = (int)(i / K), c = (int)(i % K);
  float v = (c < K1) ? a[(long)r * K1 + c] : s * b[(long)r * K2 + (c - K1)];
  dst[i] = (__bf16)v;
}

__global__ void k_cvt_strided(__bf16* __restrict__ dst, int dstride, int coloff,
                              const float* __restrict__ src, long rows, int cols) {
  long total = rows * cols;
  long i = (long)blockIdx.x * blockDim.x + threadIdx.x;
  if (i >= total) return;
  long r = i / cols; int c = (int)(i % cols);
  dst[r * dstride + coloff + c] = (__bf16)src[i];
}

// ---------------------------------------------------------------------------
// bf16 WMMA GEMM:  C[M,N] (fp32) = A[M,K] @ W[N,K]^T  (torch Linear layout)
//   block: 256 threads (8 waves), tile 128x128, K-step 32
//   double-buffered LDS, staged with GLOBAL_LOAD_ASYNC_TO_LDS_B128,
//   one workgroup barrier per K-step.
//   wave(wm 0..3, wn 0..1) computes 32x64 => 2x4 wmma accumulators
// ---------------------------------------------------------------------------
#define BM 128
#define BN 128
#define BK 32
#define LDSS 40   // padded bf16 row stride

__device__ __forceinline__ v16bf mk16(v8bf lo, v8bf hi) {
  v16bf r;
#pragma unroll
  for (int i = 0; i < 8; i++) { r[i] = lo[i]; r[8 + i] = hi[i]; }
  return r;
}

__global__ __launch_bounds__(256) void gemm_bf16(
    const __bf16* __restrict__ A, int lda,
    const __bf16* __restrict__ W,          // [N,K] contiguous
    float* __restrict__ C, int ldc,
    int M, int N, int K, int relu_ep) {
  __shared__ __bf16 sA[2][BM * LDSS];
  __shared__ __bf16 sB[2][BN * LDSS];

  int nblk = N / BN;
  int bm = blockIdx.x / nblk;
  int bn = blockIdx.x % nblk;
  int tid = threadIdx.x;
  int lane = tid & 31;
  int wave = tid >> 5;
  int wm = wave & 3;        // M offset = wm*32
  int wn = wave >> 2;       // N offset = wn*64

  int l  = lane & 15;
  int hi = lane >> 4;

  v8f acc[2][4];
#pragma unroll
  for (int mi = 0; mi < 2; mi++)
#pragma unroll
    for (int ni = 0; ni < 4; ni++) acc[mi][ni] = (v8f){0.f,0.f,0.f,0.f,0.f,0.f,0.f,0.f};

  const __bf16* Ab = A + (size_t)bm * BM * lda;
  const __bf16* Wb = W + (size_t)bn * BN * K;

  int lr = tid >> 1;          // 0..127 tile row
  int lc = (tid & 1) * 16;    // 0 or 16 (bf16 elems)

  const __bf16* Arow = Ab + (size_t)lr * lda + lc;
  const __bf16* Wrow = Wb + (size_t)lr * K  + lc;
  uint32_t ldsA0 = (uint32_t)(uintptr_t)&sA[0][lr * LDSS + lc];
  uint32_t ldsA1 = (uint32_t)(uintptr_t)&sA[1][lr * LDSS + lc];
  uint32_t ldsB0 = (uint32_t)(uintptr_t)&sB[0][lr * LDSS + lc];
  uint32_t ldsB1 = (uint32_t)(uintptr_t)&sB[1][lr * LDSS + lc];

  // each thread stages 16 bf16 (two b128 async copies) of A and of B per tile
  auto issue = [&](int buf, int k0) {
    uint32_t la = buf ? ldsA1 : ldsA0;
    uint32_t lb = buf ? ldsB1 : ldsB0;
    async_copy_b128(la,      Arow + k0);
    async_copy_b128(la + 16, Arow + k0 + 8);
    async_copy_b128(lb,      Wrow + k0);
    async_copy_b128(lb + 16, Wrow + k0 + 8);
  };

  issue(0, 0);
  wait_asynccnt0();
  __syncthreads();

  int cur = 0;
  for (int k0 = 0; k0 < K; k0 += BK) {
    int nk = k0 + BK;
    if (nk < K) issue(cur ^ 1, nk);   // prefetch next tile into other buffer

    const __bf16* cA = sA[cur];
    const __bf16* cB = sB[cur];

    // A fragments: lanes 0-15 -> rows m, K {0..7,16..23}; lanes 16-31 -> K {8..15,24..31}
    v16bf afrag[2];
#pragma unroll
    for (int mi = 0; mi < 2; mi++) {
      int row = wm * 32 + mi * 16 + l;
      int ka = hi ? 8 : 0;
      v8bf a0 = *(const v8bf*)&cA[row * LDSS + ka];
      v8bf a1 = *(const v8bf*)&cA[row * LDSS + ka + 16];
      afrag[mi] = mk16(a0, a1);
    }
    // B fragments: lane col n = l, lanes 0-15 K {0..7,16..23}, lanes 16-31 K {8..15,24..31}
    v16bf bfrag[4];
#pragma unroll
    for (int ni = 0; ni < 4; ni++) {
      int col = wn * 64 + ni * 16 + l;
      int kb = hi * 8;
      v8bf b0 = *(const v8bf*)&cB[col * LDSS + kb];
      v8bf b1 = *(const v8bf*)&cB[col * LDSS + kb + 16];
      bfrag[ni] = mk16(b0, b1);
    }

#pragma unroll
    for (int mi = 0; mi < 2; mi++)
#pragma unroll
      for (int ni = 0; ni < 4; ni++)
        acc[mi][ni] = __builtin_amdgcn_wmma_f32_16x16x32_bf16(
            false, afrag[mi], false, bfrag[ni], (short)0, acc[mi][ni], false, false);

    wait_asynccnt0();    // my prefetches for the other buffer are done
    __syncthreads();     // everyone's prefetches done + everyone done reading cur
    cur ^= 1;
  }

  // epilogue: C layout — lanes 0-15: M=g, N=l ; lanes 16-31: M=g+8, N=l
#pragma unroll
  for (int mi = 0; mi < 2; mi++) {
    int mrow0 = bm * BM + wm * 32 + mi * 16 + hi * 8;
#pragma unroll
    for (int ni = 0; ni < 4; ni++) {
      int col = bn * BN + wn * 64 + ni * 16 + l;
#pragma unroll
      for (int g = 0; g < 8; g++) {
        float v = acc[mi][ni][g];
        if (relu_ep) v = fmaxf(v, 0.f);
        C[(size_t)(mrow0 + g) * ldc + col] = v;
      }
    }
  }
}

// ---------------------------------------------------------------------------
// elementwise stages  (i over B*L, row = i>>9, col = i&511)
// ---------------------------------------------------------------------------
__global__ void k_ew1(float* __restrict__ out,
                      const float* __restrict__ sigmap_h,   // tmpL
                      const float* __restrict__ C1,         // [B,1536] mu_q|sigma_q|it_theta
                      const float* __restrict__ spp,
                      const float* __restrict__ tfp,
                      const float* __restrict__ epsz,
                      const float* __restrict__ bps,
                      __bf16* __restrict__ spbf,
                      float* __restrict__ rawz) {
  long i = (long)blockIdx.x * blockDim.x + threadIdx.x;
  if (i >= (long)B_ROWS * L_DIM) return;
  long row = i >> 9; int col = (int)(i & 511);
  long ro = row * OUT_W;
  float sp = 0.8f * fmaxf(sigmap_h[i] + fmaxf(bps[col], 0.f), 0.f) + 0.2f * spp[i];
  out[ro + O_SP + col] = sp;
  spbf[i] = (__bf16)sp;
  float itth = C1[row * 1536 + 1024 + col];
  float t0 = tfp[i];
  float tf = 0.4f * t0 + expf(-50.f * fabsf(t0)) * itth;
  float t = tanhf(tf); t = t * t;
  out[ro + O_TFF + col] = t;
  float mu_q = fmaxf(C1[row * 1536 + col], 0.f);
  float sq   = fmaxf(C1[row * 1536 + 512 + col], 0.f);
  float rz = mu_q + epsz[i] * (1.f / (1.f + expf(-0.01f * sq)) - 0.5f);
  rawz[i] = fmaxf(tanhf(rz), 0.f);
}

__global__ void k_ew2(float* __restrict__ out,
                      const float* __restrict__ vip,        // tmpL
                      const float* __restrict__ thp,
                      __bf16* __restrict__ thbf) {
  long i = (long)blockIdx.x * blockDim.x + threadIdx.x;
  if (i >= (long)B_ROWS * L_DIM) return;
  long row = i >> 9; int col = (int)(i & 511);
  long ro = row * OUT_W;
  float tff = out[ro + O_TFF + col];
  float th = 0.1f * thp[i] + tff / (1.f + vip[i]);
  out[ro + O_TH + col] = th;
  thbf[i] = (__bf16)th;
}

__global__ void k_ew3(float* __restrict__ out,
                      const float* __restrict__ t2z,        // tmpL
                      const float* __restrict__ sstp,
                      const float* __restrict__ rawz,
                      const float* __restrict__ mu_p,
                      const float* __restrict__ epszh,
                      __bf16* __restrict__ zbf,
                      __bf16* __restrict__ Azh,
                      __bf16* __restrict__ Azh2) {
  long i = (long)blockIdx.x * blockDim.x + threadIdx.x;
  if (i >= (long)B_ROWS * L_DIM) return;
  long row = i >> 9; int col = (int)(i & 511);
  long ro = row * OUT_W;
  float sst = 0.8f * sstp[i] + t2z[i];
  out[ro + O_SST + col] = sst;
  float z = fmaxf(rawz[i] - sst, 0.f);
  out[ro + O_Z + col] = z;
  out[ro + O_ZE + col] = z;       // z_energy == z in forward
  __bf16 zb = (__bf16)z;
  zbf[i] = zb;
  Azh[row * 1024 + col] = zb;
  Azh2[row * 1024 + col] = zb;
  float sp = out[ro + O_SP + col];
  float zh = mu_p[i] + epszh[i] * sp;
  float d = z - zh;
  out[ro + O_L2 + col] = d * d;
}

__global__ void k_ew5(float* __restrict__ out,
                      const float* __restrict__ r2,         // [B,1024]
                      const float* __restrict__ I_t) {
  long i = (long)blockIdx.x * blockDim.x + threadIdx.x;
  if (i >= (long)B_ROWS * D_DIM) return;
  long row = i >> 10; int col = (int)(i & 1023);
  long ro = row * OUT_W;
  float ih = 1.f / (1.f + expf(-(r2[i] - 2.f)));
  out[ro + O_IHAT + col] = ih;
  float d = I_t[i] - ih;
  out[ro + O_L1 + col] = d * d;
}

// ---------------------------------------------------------------------------
// launch
// ---------------------------------------------------------------------------
extern "C" void kernel_launch(void* const* d_in, const int* in_sizes, int n_in,
                              void* d_out, int out_size, void* d_ws, size_t ws_size,
                              hipStream_t stream) {
  const float* I_t        = (const float*)d_in[0];
  const float* h          = (const float*)d_in[1];
  const float* h2         = (const float*)d_in[2];
  const float* spp        = (const float*)d_in[3];
  const float* tfp        = (const float*)d_in[4];
  const float* thp        = (const float*)d_in[5];
  const float* sstp       = (const float*)d_in[6];
  const float* epsz       = (const float*)d_in[7];
  const float* epszh      = (const float*)d_in[8];
  const float* W_post_mu  = (const float*)d_in[9];
  const float* W_post_sg  = (const float*)d_in[10];
  const float* W_z_to_h   = (const float*)d_in[11];
  const float* W_h_to_h   = (const float*)d_in[12];
  const float* W_h2_to_h2 = (const float*)d_in[13];
  const float* W_z_to_h2  = (const float*)d_in[14];
  const float* W_prior_mu = (const float*)d_in[15];
  const float* W_prior_sg = (const float*)d_in[16];
  const float* b_prior_sg = (const float*)d_in[17];
  const float* W_vip      = (const float*)d_in[18];
  const float* W_t2z      = (const float*)d_in[19];
  const float* W_I2t      = (const float*)d_in[20];
  const float* W_rec1     = (const float*)d_in[21];
  const float* W_rec2     = (const float*)d_in[22];
  float* out = (float*)d_out;

  char* ws = (char*)d_ws;
  size_t off = 0;
  auto alloc = [&](size_t bytes) { size_t r = off; off += (bytes + 255) & ~(size_t)255; return r; };

  const size_t Bn = B_ROWS;
  float*  hdr    = (float*) (ws + alloc(256));
  __bf16* WcatI  = (__bf16*)(ws + alloc((size_t)1536 * 1024 * 2));
  __bf16* Wpm    = (__bf16*)(ws + alloc((size_t)512 * 512 * 2));
  __bf16* Wps    = (__bf16*)(ws + alloc((size_t)512 * 512 * 2));
  __bf16* Wvip   = (__bf16*)(ws + alloc((size_t)512 * 512 * 2));
  __bf16* Wt2z   = (__bf16*)(ws + alloc((size_t)512 * 512 * 2));
  __bf16* Wcath  = (__bf16*)(ws + alloc((size_t)512 * 1024 * 2));
  __bf16* Wcath2 = (__bf16*)(ws + alloc((size_t)512 * 1024 * 2));
  __bf16* Wr1    = (__bf16*)(ws + alloc((size_t)256 * 512 * 2));
  __bf16* Wr2    = (__bf16*)(ws + alloc((size_t)1024 * 256 * 2));
  __bf16* Ibf    = (__bf16*)(ws + alloc(Bn * 1024 * 2));
  __bf16* Azh    = (__bf16*)(ws + alloc(Bn * 1024 * 2));
  __bf16* Azh2   = (__bf16*)(ws + alloc(Bn * 1024 * 2));
  __bf16* spbf   = (__bf16*)(ws + alloc(Bn * 512 * 2));
  __bf16* thbf   = (__bf16*)(ws + alloc(Bn * 512 * 2));
  __bf16* zbf    = (__bf16*)(ws + alloc(Bn * 512 * 2));
  __bf16* r1bf   = (__bf16*)(ws + alloc(Bn * 256 * 2));
  float*  C1     = (float*) (ws + alloc(Bn * 1536 * 4));  // mu_q|sigma_q|it_theta ; later aliased as r2
  float*  mu_p   = (float*) (ws + alloc(Bn * 512 * 4));
  float*  tmpL   = (float*) (ws + alloc(Bn * 512 * 4));   // sigmap_h -> vip -> t2z
  float*  rawz   = (float*) (ws + alloc(Bn * 512 * 4));
  float*  r1f    = (float*) (ws + alloc(Bn * 256 * 4));
  float*  r2f    = C1;                                    // alias (C1 dead after ew1)

  auto cdiv = [](long a, long b) { return (unsigned)((a + b - 1) / b); };
  auto gemm = [&](const __bf16* Ap, int lda, const __bf16* Wp, float* Cp, int ldc,
                  int M, int N, int K, int ep) {
    gemm_bf16<<<dim3((M / 128) * (N / 128)), dim3(256), 0, stream>>>(Ap, lda, Wp, Cp, ldc, M, N, K, ep);
  };

  // 1) ||W_h_to_h||_F^2
  k_zero<<<1, 64, 0, stream>>>(hdr, 64);
  k_sumsq<<<256, 256, 0, stream>>>(W_h_to_h, 512 * 512, hdr);

  // 2) pack weights to bf16
  k_cvt<<<cdiv(512 * 1024, 256), 256, 0, stream>>>(WcatI,               W_post_mu, 512 * 1024);
  k_cvt<<<cdiv(512 * 1024, 256), 256, 0, stream>>>(WcatI + 512 * 1024,  W_post_sg, 512 * 1024);
  k_cvt<<<cdiv(512 * 1024, 256), 256, 0, stream>>>(WcatI + 1024 * 1024, W_I2t,     512 * 1024);
  k_cvt<<<cdiv(512 * 512, 256), 256, 0, stream>>>(Wpm, W_prior_mu, 512 * 512);
  k_cvt<<<cdiv(512 * 512, 256), 256, 0, stream>>>(Wps, W_prior_sg, 512 * 512);
  k_cvt_relu<<<cdiv(512 * 512, 256), 256, 0, stream>>>(Wvip, W_vip, 512 * 512);
  k_cvt_relu<<<cdiv(512 * 512, 256), 256, 0, stream>>>(Wt2z, W_t2z, 512 * 512);
  k_wcat<<<cdiv(512 * 1024, 256), 256, 0, stream>>>(Wcath,  W_z_to_h,  W_h_to_h,   512, 512, 512, hdr);
  k_wcat<<<cdiv(512 * 1024, 256), 256, 0, stream>>>(Wcath2, W_z_to_h2, W_h2_to_h2, 512, 512, 512, nullptr);
  k_cvt<<<cdiv(256 * 512, 256), 256, 0, stream>>>(Wr1, W_rec1, 256 * 512);
  k_cvt<<<cdiv(1024 * 256, 256), 256, 0, stream>>>(Wr2, W_rec2, 1024 * 256);

  // 3) pack activations
  k_cvt<<<cdiv(Bn * 1024, 256), 256, 0, stream>>>(Ibf, I_t, Bn * 1024);
  k_cvt_strided<<<cdiv(Bn * 512, 256), 256, 0, stream>>>(Azh,  1024, 512, h,  Bn, 512);
  k_cvt_strided<<<cdiv(Bn * 512, 256), 256, 0, stream>>>(Azh2, 1024, 512, h2, Bn, 512);

  // 4) input-side GEMMs
  gemm(Ibf, 1024, WcatI, C1, 1536, B_ROWS, 1536, 1024, 0);          // mu_q|sigma_q|it_theta (raw)
  gemm(Azh2 + 512, 1024, Wpm, mu_p, 512, B_ROWS, 512, 512, 1);      // mu_p = relu(h2 Wpm^T)
  gemm(Azh  + 512, 1024, Wps, tmpL, 512, B_ROWS, 512, 512, 0);      // sigmap_h (pre-bias)

  // 5) sigma_p / theta_ff / raw_z
  k_ew1<<<cdiv(Bn * 512, 256), 256, 0, stream>>>(out, tmpL, C1, spp, tfp, epsz, b_prior_sg, spbf, rawz);

  // 6) VIP chain
  gemm(spbf, 512, Wvip, tmpL, 512, B_ROWS, 512, 512, 0);            // vip_inh
  k_ew2<<<cdiv(Bn * 512, 256), 256, 0, stream>>>(out, tmpL, thp, thbf);
  gemm(thbf, 512, Wt2z, tmpL, 512, B_ROWS, 512, 512, 0);            // theta @ W_t2z^T
  k_ew3<<<cdiv(Bn * 512, 256), 256, 0, stream>>>(out, tmpL, sstp, rawz, mu_p, epszh, zbf, Azh, Azh2);

  // 7) reconstruction
  gemm(zbf, 512, Wr1, r1f, 256, B_ROWS, 256, 512, 0);
  k_cvt<<<cdiv(Bn * 256, 256), 256, 0, stream>>>(r1bf, r1f, Bn * 256);
  gemm(r1bf, 256, Wr2, r2f, 1024, B_ROWS, 1024, 256, 0);
  k_ew5<<<cdiv(Bn * 1024, 256), 256, 0, stream>>>(out, r2f, I_t);

  // 8) recurrent state updates, fused relu, write straight into strided out
  gemm(Azh,  1024, Wcath,  out + O_HN,  OUT_W, B_ROWS, 512, 1024, 1);  // h_new
  gemm(Azh2, 1024, Wcath2, out + O_H2N, OUT_W, B_ROWS, 512, 1024, 1);  // h2_new
}
